// SpiralEnblock_2808908611872
// MI455X (gfx1250) — matrix operand order repaired and checked by reading, not compiled, not run on hardware
//
#include <hip/hip_runtime.h>

typedef __attribute__((ext_vector_type(16))) _Float16 v16h;
typedef __attribute__((ext_vector_type(8)))  float    v8f;
typedef __attribute__((ext_vector_type(4)))  float    v4f;
typedef __attribute__((ext_vector_type(4)))  int      v4i;

#define BS    16
#define NN    20000
#define CIN   32
#define SEQ   9
#define COUT  64
#define MPOOL 5000
#define NNZ   20000
#define FAN   288   /* SEQ * CIN */

#if defined(__has_builtin)
#if __has_builtin(__builtin_amdgcn_global_load_async_to_lds_b128)
#define HAVE_ASYNC_LDS 1
#endif
#endif

// ---------------------------------------------------------------------------
// Kernel 1: fused gather + GEMM (f16 WMMA, f32 accum) + bias + ELU -> h in ws
// block = 128 threads (4 waves). Block owns one 16-node tile; wave `w` keeps
// batches 4w..4w+3 *concurrently* (16 v8f accumulators) so the 4 B-fragments
// per K-step are reused across batches without LICM hoisting 36 fragments
// into VGPRs. h layout: [BS][NN][COUT] fp32.
// ---------------------------------------------------------------------------
__global__ __launch_bounds__(128)
void spiral_gemm_kernel(const float* __restrict__ x,
                        const int*   __restrict__ indices,
                        const float* __restrict__ weight,
                        const float* __restrict__ bias,
                        float*       __restrict__ h_out) {
  __shared__ float wlds[COUT * FAN];   // 72 KB: weight fp32, row-major [o][k]

  const int tid = threadIdx.x;

  // ---- stage weight into LDS (async DMA path when available) ----
#ifdef HAVE_ASYNC_LDS
  {
    // builtin signature: (v4i addrspace(1)*, v4i addrspace(3)*, imm offset, imm cpol)
    __attribute__((address_space(1))) v4i* gsrc =
        (__attribute__((address_space(1))) v4i*)((v4i*)weight);
    __attribute__((address_space(3))) v4i* ldst =
        (__attribute__((address_space(3))) v4i*)((v4i*)wlds);
    // 64*288*4 B = 4608 x 16-byte chunks; 36 per thread
#pragma unroll 1
    for (int j = 0; j < 36; ++j) {
      const int c = tid + 128 * j;
      __builtin_amdgcn_global_load_async_to_lds_b128(gsrc + c, ldst + c, 0, 0);
    }
  }
#if __has_builtin(__builtin_amdgcn_s_wait_asynccnt)
  __builtin_amdgcn_s_wait_asynccnt(0);
#else
  asm volatile("s_wait_asynccnt 0x0" ::: "memory");
#endif
#else
  for (int i = tid; i < COUT * FAN; i += 128) wlds[i] = weight[i];
#endif
  __syncthreads();

  const int lane = tid & 31;
  const int wave = tid >> 5;
  const int hi   = lane >> 4;        // half-wave select
  const int l16  = lane & 15;
  const int n0   = blockIdx.x * 16;  // node tile base (1250 tiles exactly)
  const int b0   = wave * 4;         // this wave's 4 batches

  // spiral neighbor indices for this lane's A-matrix row (M = l16)
  int nidx[SEQ];
#pragma unroll
  for (int s = 0; s < SEQ; ++s)
    nidx[s] = indices[(n0 + l16) * SEQ + s];

  // bias: C element (M,N) depends only on N = nt*16 + l16
  float bvals[4];
#pragma unroll
  for (int nt = 0; nt < 4; ++nt)
    bvals[nt] = bias[nt * 16 + l16];

  const int cbase = hi ? 16 : 0;     // A layout: hi half-wave owns K+16..K+31

  v8f acc[4][4];                     // [batch][n-tile]
#pragma unroll
  for (int b4 = 0; b4 < 4; ++b4)
#pragma unroll
    for (int nt = 0; nt < 4; ++nt) {
      v8f t;
#pragma unroll
      for (int i = 0; i < 8; ++i) t[i] = bvals[nt];
      acc[b4][nt] = t;
    }

#pragma unroll 1
  for (int s = 0; s < SEQ; ++s) {
    // prefetch next spiral row for all 4 batches (global_prefetch_b8)
    if (s + 1 < SEQ) {
      const int nn = nidx[s + 1];
#pragma unroll
      for (int b4 = 0; b4 < 4; ++b4)
        __builtin_prefetch(x + ((size_t)(b0 + b4) * NN + nn) * CIN, 0, 3);
    }

    // ---- B fragments for this K-step (shared by all 4 batches) ----
    // B k order used here (contiguous, same as A): lane = column N, halves = K, hi half-wave = K+16
    const int kb = s * 32 + (hi ? 16 : 0);
    v16h bf[4];
#pragma unroll
    for (int nt = 0; nt < 4; ++nt) {
      const float* wp = &wlds[(nt * 16 + l16) * FAN + kb];
      v4f w0 = *(const v4f*)(wp + 0);
      v4f w1 = *(const v4f*)(wp + 4);
      v4f w2 = *(const v4f*)(wp + 8);
      v4f w3 = *(const v4f*)(wp + 12);
      v16h t;
#pragma unroll
      for (int i = 0; i < 4; ++i) {
        t[i]      = (_Float16)w0[i];
        t[4 + i]  = (_Float16)w1[i];
        t[8 + i]  = (_Float16)w2[i];
        t[12 + i] = (_Float16)w3[i];
      }
      bf[nt] = t;
    }

    // ---- per-batch gathered A fragment + 4 WMMAs ----
    const int node = nidx[s];
#pragma unroll
    for (int b4 = 0; b4 < 4; ++b4) {
      const float* xr = x + ((size_t)(b0 + b4) * NN + node) * CIN;
      // A k order used here (contiguous, same as B):
      // halves 0..15 = K cbase..cbase+15
      v4f f0 = *(const v4f*)(xr + cbase + 0);
      v4f f1 = *(const v4f*)(xr + cbase + 4);
      v4f f2 = *(const v4f*)(xr + cbase + 8);
      v4f f3 = *(const v4f*)(xr + cbase + 12);
      v16h a;
#pragma unroll
      for (int i = 0; i < 4; ++i) {
        a[i]      = (_Float16)f0[i];
        a[4 + i]  = (_Float16)f1[i];
        a[8 + i]  = (_Float16)f2[i];
        a[12 + i] = (_Float16)f3[i];
      }
#pragma unroll
      for (int nt = 0; nt < 4; ++nt) {
        acc[b4][nt] = __builtin_amdgcn_wmma_f32_16x16x32_f16(
            /*neg_a=*/false, a, /*neg_b=*/false, bf[nt],
            /*c_mod=*/(short)0, acc[b4][nt],
            /*reuse_a=*/false, /*reuse_b=*/false);
      }
    }
  }

  // ---- epilogue: ELU + store. C/D layout: VGPR v -> M = v + 8*hi, N = l16.
#pragma unroll
  for (int b4 = 0; b4 < 4; ++b4) {
    const int b = b0 + b4;
#pragma unroll
    for (int nt = 0; nt < 4; ++nt) {
      const int o = nt * 16 + l16;
#pragma unroll
      for (int v = 0; v < 8; ++v) {
        const int m = v + (hi ? 8 : 0);
        float hv = acc[b4][nt][v];
        hv = hv > 0.0f ? hv : (__expf(hv) - 1.0f);
        h_out[((size_t)b * NN + (n0 + m)) * COUT + o] = hv;
      }
    }
  }
}

// ---------------------------------------------------------------------------
// Kernel 0: zero the pooled output (graph-capture-safe, deterministic)
// ---------------------------------------------------------------------------
__global__ void zero_kernel(float* __restrict__ p, int n) {
  int i = blockIdx.x * blockDim.x + threadIdx.x;
  if (i < n) p[i] = 0.0f;
}

// ---------------------------------------------------------------------------
// Kernel 2: COO scatter-add pool. 16 threads per (k,b) entry, 4 channels each
// -> coalesced h reads, hardware f32 atomics on out.
// ---------------------------------------------------------------------------
__global__ __launch_bounds__(256)
void pool_scatter_kernel(const float* __restrict__ h,
                         const int*   __restrict__ rows,
                         const int*   __restrict__ cols,
                         const float* __restrict__ vals,
                         float*       __restrict__ out) {
  const int t  = blockIdx.x * blockDim.x + threadIdx.x;
  const int ci = t & 15;
  const int kb = t >> 4;
  if (kb >= NNZ * BS) return;
  const int k = kb % NNZ;
  const int b = kb / NNZ;

  const int   r = rows[k];
  const int   c = cols[k];
  const float v = vals[k];

  const float* hp = h   + ((size_t)b * NN    + c) * COUT + ci * 4;
  float*       op = out + ((size_t)b * MPOOL + r) * COUT + ci * 4;

  v4f hv = *(const v4f*)hp;
#pragma unroll
  for (int i = 0; i < 4; ++i)
    unsafeAtomicAdd(op + i, hv[i] * v);   // global_atomic_add_f32
}

// ---------------------------------------------------------------------------
extern "C" void kernel_launch(void* const* d_in, const int* in_sizes, int n_in,
                              void* d_out, int out_size, void* d_ws, size_t ws_size,
                              hipStream_t stream) {
  const float* x      = (const float*)d_in[0];
  const int*   idx    = (const int*)  d_in[1];
  const float* weight = (const float*)d_in[2];
  const float* bias   = (const float*)d_in[3];
  const int*   trows  = (const int*)  d_in[4];
  const int*   tcols  = (const int*)  d_in[5];
  const float* tvals  = (const float*)d_in[6];

  float* out  = (float*)d_out;
  float* hbuf = (float*)d_ws;   // BS*NN*COUT fp32 = ~82 MB, fully overwritten

  zero_kernel<<<(out_size + 255) / 256, 256, 0, stream>>>(out, out_size);
  spiral_gemm_kernel<<<NN / 16, 128, 0, stream>>>(x, idx, weight, bias, hbuf);
  pool_scatter_kernel<<<(NNZ * BS * 16 + 255) / 256, 256, 0, stream>>>(
      hbuf, trows, tcols, tvals, out);
}
